// epipolar_Attention_Block_28862180229258
// MI455X (gfx1250) — compile-verified, matrix operand order checked by reading, not run on hardware
//
#include <hip/hip_runtime.h>
#include <hip/hip_bf16.h>

// Shapes from the reference
#define BB   4
#define QC   512
#define KC   256
#define RESD 64
#define SRCD 128
#define TT   (RESD * RESD)     // 4096
#define GROUPS 32
#define CPG  (QC / GROUPS)     // 16
#define CAT  (2 * QC)          // 1024
#define SCALE 0.21022410381342863f  // 512^-0.25

typedef __attribute__((ext_vector_type(16))) __bf16 bf16x16;
typedef __attribute__((ext_vector_type(8)))  __bf16 bf16x8;
typedef __attribute__((ext_vector_type(8)))  float  v8f;

// ---------------- WMMA helpers ----------------
// Fragment lane mapping (A and B identical): lane L -> idx = L%16,
// kb = (L/16)*8; vector elems [0..7] = K kb..kb+7, [8..15] = K kb+16..kb+23.
// With operands stored row-major along K, a fragment is two 16-byte loads.

__device__ __forceinline__ bf16x16 frag_ld(const __bf16* p) {
  bf16x8 lo = *(const bf16x8*)(p);
  bf16x8 hi = *(const bf16x8*)(p + 16);
  return __builtin_shufflevector(lo, hi, 0, 1, 2, 3, 4, 5, 6, 7,
                                 8, 9, 10, 11, 12, 13, 14, 15);
}

__device__ __forceinline__ v8f wmma_bf16(bf16x16 a, bf16x16 b, v8f c) {
  return __builtin_amdgcn_wmma_f32_16x16x32_bf16(false, a, false, b, (short)0, c,
                                                 false, false);
}

// D = A(m0..m0+31, :) x Bt(n0..n0+31, :)^T ; A row-major lda, Bt n-major ldb.
__device__ __forceinline__ void gemm_core_2x2(const __bf16* __restrict__ A, int lda,
                                              const __bf16* __restrict__ Bt, int ldb,
                                              int m0, int n0, int K, int lm, int kb,
                                              v8f acc[2][2]) {
  const __bf16* pa0 = A + (size_t)(m0 + lm) * lda + kb;
  const __bf16* pa1 = pa0 + (size_t)16 * lda;
  const __bf16* pb0 = Bt + (size_t)(n0 + lm) * ldb + kb;
  const __bf16* pb1 = pb0 + (size_t)16 * ldb;
  for (int k0 = 0; k0 < K; k0 += 32) {
    bf16x16 a0 = frag_ld(pa0 + k0);
    bf16x16 a1 = frag_ld(pa1 + k0);
    bf16x16 b0 = frag_ld(pb0 + k0);
    bf16x16 b1 = frag_ld(pb1 + k0);
    acc[0][0] = wmma_bf16(a0, b0, acc[0][0]);
    acc[0][1] = wmma_bf16(a0, b1, acc[0][1]);
    acc[1][0] = wmma_bf16(a1, b0, acc[1][0]);
    acc[1][1] = wmma_bf16(a1, b1, acc[1][1]);
  }
}

// 64x64 output per wave: 8 fragment loads per 16 WMMAs.
__device__ __forceinline__ void gemm_core_4x4(const __bf16* __restrict__ A, int lda,
                                              const __bf16* __restrict__ Bt, int ldb,
                                              int m0, int n0, int K, int lm, int kb,
                                              v8f acc[4][4]) {
  const __bf16* pa[4];
  const __bf16* pb[4];
#pragma unroll
  for (int i = 0; i < 4; ++i) {
    pa[i] = A + (size_t)(m0 + 16 * i + lm) * lda + kb;
    pb[i] = Bt + (size_t)(n0 + 16 * i + lm) * ldb + kb;
  }
  for (int k0 = 0; k0 < K; k0 += 32) {
    bf16x16 a[4], b[4];
#pragma unroll
    for (int i = 0; i < 4; ++i) {
      a[i] = frag_ld(pa[i] + k0);
      b[i] = frag_ld(pb[i] + k0);
    }
#pragma unroll
    for (int i = 0; i < 4; ++i)
#pragma unroll
      for (int j = 0; j < 4; ++j) acc[i][j] = wmma_bf16(a[i], b[j], acc[i][j]);
  }
}

// ---------------- weight f32 -> bf16 conversion ----------------
__global__ void convert_kernel(const float* __restrict__ src, __bf16* __restrict__ dst, int n) {
  int i = blockIdx.x * blockDim.x + threadIdx.x;
  if (i < n) dst[i] = (__bf16)src[i];
}

// ---------------- Stage 1: 2x2 avg pool -> kdsT[b][t][c] (bf16) --------------
__global__ void pool_kernel(const float* __restrict__ kf, __bf16* __restrict__ kdsT) {
  int idx = blockIdx.x * blockDim.x + threadIdx.x;  // over BB*TT*KC, c fastest
  if (idx >= BB * TT * KC) return;
  int c  = idx % KC;
  int bt = idx / KC;
  int t  = bt % TT;
  int b  = bt / TT;
  int x = t % RESD, y = t / RESD;
  const float* p = kf + (((size_t)b * KC + c) * SRCD + 2 * y) * SRCD + 2 * x;
  kdsT[idx] = (__bf16)(0.25f * (p[0] + p[1] + p[SRCD] + p[SRCD + 1]));
}

// ---------------- GroupNorm: src[b][c][t] -> optional f32 dst + bf16 cnT -----
__global__ void groupnorm_kernel(const float* __restrict__ src, float* __restrict__ dstf,
                                 __bf16* __restrict__ cnT, int colOff,
                                 const float* __restrict__ w, const float* __restrict__ bias) {
  int b = blockIdx.x / GROUPS;
  int g = blockIdx.x % GROUPS;
  const int N = CPG * TT;  // 65536
  const float* base = src + ((size_t)b * QC + g * CPG) * TT;

  float s = 0.f, s2 = 0.f;
  for (int i = threadIdx.x; i < N; i += blockDim.x) {
    float v = base[i];
    s += v;
    s2 += v * v;
  }
  __shared__ float rs[256], rs2[256];
  rs[threadIdx.x] = s;
  rs2[threadIdx.x] = s2;
  __syncthreads();
  for (int off = 128; off > 0; off >>= 1) {
    if ((int)threadIdx.x < off) {
      rs[threadIdx.x]  += rs[threadIdx.x + off];
      rs2[threadIdx.x] += rs2[threadIdx.x + off];
    }
    __syncthreads();
  }
  float mean = rs[0] / (float)N;
  float var  = rs2[0] / (float)N - mean * mean;
  float rstd = rsqrtf(var + 1e-5f);

  float* obase = dstf ? (dstf + ((size_t)b * QC + g * CPG) * TT) : nullptr;
  for (int i = threadIdx.x; i < N; i += blockDim.x) {
    int cl = i / TT;            // channel within group
    int t  = i - cl * TT;
    int c  = g * CPG + cl;      // global channel
    float v = (base[i] - mean) * rstd * w[c] + bias[c];
    if (obase) obase[i] = v;
    cnT[((size_t)b * TT + t) * CAT + colOff + c] = (__bf16)v;
  }
}

// ---------------- Stage 3: key projection -> kraw f32 [b][c][t] --------------
__global__ void kproj_kernel(const __bf16* __restrict__ kpw, const __bf16* __restrict__ kdsT,
                             float* __restrict__ kraw) {
  int wave = (blockIdx.x * blockDim.x + threadIdx.x) >> 5;
  int lane = threadIdx.x & 31;
  const int MT2 = QC / 32, NT2 = TT / 32;  // 16, 128
  int b  = wave / (MT2 * NT2);
  int r  = wave % (MT2 * NT2);
  int m0 = (r / NT2) * 32;
  int n0 = (r % NT2) * 32;
  int lm = lane & 15, half = lane >> 4, kb = half * 8;

  v8f acc[2][2] = {};
  gemm_core_2x2(kpw, KC, kdsT + (size_t)b * TT * KC, KC, m0, n0, KC, lm, kb, acc);

  float* outb = kraw + (size_t)b * QC * TT;
#pragma unroll
  for (int sm = 0; sm < 2; ++sm)
#pragma unroll
    for (int sn = 0; sn < 2; ++sn)
#pragma unroll
      for (int rr = 0; rr < 8; ++rr) {
        int row = m0 + sm * 16 + rr + 8 * half;
        int col = n0 + sn * 16 + lm;
        outb[(size_t)row * TT + col] = acc[sm][sn][rr];
      }
}

// ---------------- Stage 5: qkv GEMM -> qqT/kkT (scaled, [t][c]) + vv [c][t] --
__global__ void qkv_kernel(const __bf16* __restrict__ qkvw, const float* __restrict__ qkvb,
                           const __bf16* __restrict__ cnT,
                           __bf16* __restrict__ qqT, __bf16* __restrict__ kkT,
                           __bf16* __restrict__ vv) {
  int wave = (blockIdx.x * blockDim.x + threadIdx.x) >> 5;
  int lane = threadIdx.x & 31;
  const int MT2 = (3 * QC) / 32, NT2 = TT / 32;  // 48, 128
  int b  = wave / (MT2 * NT2);
  int r  = wave % (MT2 * NT2);
  int m0 = (r / NT2) * 32;
  int n0 = (r % NT2) * 32;
  int lm = lane & 15, half = lane >> 4, kb = half * 8;

  v8f acc[2][2] = {};
  gemm_core_2x2(qkvw, CAT, cnT + (size_t)b * TT * CAT, CAT, m0, n0, CAT, lm, kb, acc);

#pragma unroll
  for (int sm = 0; sm < 2; ++sm)
#pragma unroll
    for (int sn = 0; sn < 2; ++sn)
#pragma unroll
      for (int rr = 0; rr < 8; ++rr) {
        int m = m0 + sm * 16 + rr + 8 * half;
        int t = n0 + sn * 16 + lm;
        float v = acc[sm][sn][rr] + qkvb[m];
        if (m < QC) {
          qqT[((size_t)b * TT + t) * QC + m] = (__bf16)(v * SCALE);
        } else if (m < 2 * QC) {
          kkT[((size_t)b * TT + t) * QC + (m - QC)] = (__bf16)(v * SCALE);
        } else {
          vv[((size_t)b * QC + (m - 2 * QC)) * TT + t] = (__bf16)v;
        }
      }
}

// ---------------- Stage 6: logits = Wm * (qqT x kkT^T), one batch ------------
__global__ void logits_kernel(int b, const __bf16* __restrict__ qqT,
                              const __bf16* __restrict__ kkT,
                              const float* __restrict__ Wm, float* __restrict__ logits) {
  int wave = (blockIdx.x * blockDim.x + threadIdx.x) >> 5;
  int lane = threadIdx.x & 31;
  const int NT4 = TT / 64;  // 64 ; M tiles4 also 64
  int t0 = (wave / NT4) * 64;
  int s0 = (wave % NT4) * 64;
  int lm = lane & 15, half = lane >> 4, kb = half * 8;

  v8f acc[4][4] = {};
  gemm_core_4x4(qqT + (size_t)b * TT * QC, QC,
                kkT + (size_t)b * TT * QC, QC, t0, s0, QC, lm, kb, acc);

  const float* wb = Wm + (size_t)b * TT * TT;
#pragma unroll
  for (int sm = 0; sm < 4; ++sm)
#pragma unroll
    for (int sn = 0; sn < 4; ++sn)
#pragma unroll
      for (int rr = 0; rr < 8; ++rr) {
        int t = t0 + sm * 16 + rr + 8 * half;
        int s = s0 + sn * 16 + lm;
        size_t idx = (size_t)t * TT + s;
        logits[idx] = acc[sm][sn][rr] * wb[idx];
      }
}

// ---------------- Stage 7: row softmax, f32 logits -> bf16 probs [t][s] ------
__global__ void softmax_kernel(const float* __restrict__ logits, __bf16* __restrict__ probs) {
  int t = blockIdx.x;
  const float* row = logits + (size_t)t * TT;
  __shared__ float red[256];

  float m = -1e30f;
  for (int i = threadIdx.x; i < TT; i += blockDim.x) m = fmaxf(m, row[i]);
  red[threadIdx.x] = m;
  __syncthreads();
  for (int off = 128; off > 0; off >>= 1) {
    if ((int)threadIdx.x < off)
      red[threadIdx.x] = fmaxf(red[threadIdx.x], red[threadIdx.x + off]);
    __syncthreads();
  }
  float rowmax = red[0];
  __syncthreads();

  float s = 0.f;
  for (int i = threadIdx.x; i < TT; i += blockDim.x) s += __expf(row[i] - rowmax);
  red[threadIdx.x] = s;
  __syncthreads();
  for (int off = 128; off > 0; off >>= 1) {
    if ((int)threadIdx.x < off) red[threadIdx.x] += red[threadIdx.x + off];
    __syncthreads();
  }
  float inv = 1.f / red[0];

  for (int i = threadIdx.x; i < TT; i += blockDim.x) {
    probs[(size_t)t * TT + i] = (__bf16)(__expf(row[i] - rowmax) * inv);
  }
}

// ---------------- Stage 8: a = vv x probs^T -> abfT[t][c], one batch ---------
// D[m=c][n=t] = sum_s vv[c][s] * probs[t][s]  (both operands contiguous in K)
__global__ void av_kernel(int b, const __bf16* __restrict__ vv,
                          const __bf16* __restrict__ probs, __bf16* __restrict__ abfT) {
  int wave = (blockIdx.x * blockDim.x + threadIdx.x) >> 5;
  int lane = threadIdx.x & 31;
  const int NT4 = TT / 64;  // 64 t-tiles4; M tiles4 = QC/64 = 8
  int c0 = (wave / NT4) * 64;
  int t0 = (wave % NT4) * 64;
  int lm = lane & 15, half = lane >> 4, kb = half * 8;

  v8f acc[4][4] = {};
  gemm_core_4x4(vv + (size_t)b * QC * TT, TT, probs, TT, c0, t0, TT, lm, kb, acc);

#pragma unroll
  for (int sm = 0; sm < 4; ++sm)
#pragma unroll
    for (int sn = 0; sn < 4; ++sn)
#pragma unroll
      for (int rr = 0; rr < 8; ++rr) {
        int c = c0 + sm * 16 + rr + 8 * half;
        int t = t0 + sn * 16 + lm;
        abfT[((size_t)b * TT + t) * QC + c] = (__bf16)acc[sm][sn][rr];
      }
}

// ---------------- Stage 9: out = qn + proj_w @ a + proj_b --------------------
__global__ void proj_kernel(const __bf16* __restrict__ pw, const float* __restrict__ pb,
                            const __bf16* __restrict__ abfT, const float* __restrict__ qn,
                            float* __restrict__ out) {
  int wave = (blockIdx.x * blockDim.x + threadIdx.x) >> 5;
  int lane = threadIdx.x & 31;
  const int MT2 = QC / 32, NT2 = TT / 32;  // 16, 128
  int b  = wave / (MT2 * NT2);
  int r  = wave % (MT2 * NT2);
  int m0 = (r / NT2) * 32;
  int n0 = (r % NT2) * 32;
  int lm = lane & 15, half = lane >> 4, kb = half * 8;

  v8f acc[2][2] = {};
  gemm_core_2x2(pw, QC, abfT + (size_t)b * TT * QC, QC, m0, n0, QC, lm, kb, acc);

#pragma unroll
  for (int sm = 0; sm < 2; ++sm)
#pragma unroll
    for (int sn = 0; sn < 2; ++sn)
#pragma unroll
      for (int rr = 0; rr < 8; ++rr) {
        int o = m0 + sm * 16 + rr + 8 * half;
        int t = n0 + sn * 16 + lm;
        size_t idx = ((size_t)b * QC + o) * TT + t;
        out[idx] = qn[idx] + acc[sm][sn][rr] + pb[o];
      }
}

// ---------------- Host launcher ----------------------------------------------
extern "C" void kernel_launch(void* const* d_in, const int* in_sizes, int n_in,
                              void* d_out, int out_size, void* d_ws, size_t ws_size,
                              hipStream_t stream) {
  const float* key_feat = (const float*)d_in[0];
  const float* query    = (const float*)d_in[1];
  const float* Wm       = (const float*)d_in[2];
  const float* kpw      = (const float*)d_in[3];
  const float* norm_w   = (const float*)d_in[4];
  const float* norm_b   = (const float*)d_in[5];
  const float* qkvw     = (const float*)d_in[6];
  const float* qkvb     = (const float*)d_in[7];
  const float* pw       = (const float*)d_in[8];
  const float* pb       = (const float*)d_in[9];
  float* out = (float*)d_out;

  char* ws = (char*)d_ws;
  size_t off = 0;
  __bf16* kpw_bf  = (__bf16*)(ws + off); off += (size_t)QC * KC * 2;
  __bf16* qkvw_bf = (__bf16*)(ws + off); off += (size_t)(3 * QC) * CAT * 2;
  __bf16* pw_bf   = (__bf16*)(ws + off); off += (size_t)QC * QC * 2;
  __bf16* kdsT    = (__bf16*)(ws + off); off += (size_t)BB * TT * KC * 2;
  float*  qn      = (float*)(ws + off);  off += (size_t)BB * QC * TT * 4;
  float*  kraw    = (float*)(ws + off);  off += (size_t)BB * QC * TT * 4;
  __bf16* cnT     = (__bf16*)(ws + off); off += (size_t)BB * TT * CAT * 2;
  __bf16* qqT     = (__bf16*)(ws + off); off += (size_t)BB * TT * QC * 2;
  __bf16* kkT     = (__bf16*)(ws + off); off += (size_t)BB * TT * QC * 2;
  __bf16* vv      = (__bf16*)(ws + off); off += (size_t)BB * QC * TT * 2;
  __bf16* abfT    = (__bf16*)(ws + off); off += (size_t)BB * TT * QC * 2;
  float*  logits  = (float*)(ws + off);  off += (size_t)TT * TT * 4;   // 1 batch
  __bf16* probs   = (__bf16*)(ws + off); off += (size_t)TT * TT * 2;   // 1 batch

  // 0. weight conversion to bf16
  {
    int n = QC * KC;
    convert_kernel<<<(n + 255) / 256, 256, 0, stream>>>(kpw, kpw_bf, n);
    n = 3 * QC * CAT;
    convert_kernel<<<(n + 255) / 256, 256, 0, stream>>>(qkvw, qkvw_bf, n);
    n = QC * QC;
    convert_kernel<<<(n + 255) / 256, 256, 0, stream>>>(pw, pw_bf, n);
  }
  // 1. 2x2 average pool -> kdsT
  {
    int n = BB * TT * KC;
    pool_kernel<<<(n + 255) / 256, 256, 0, stream>>>(key_feat, kdsT);
  }
  // 2. GroupNorm(query) -> qn f32 + cnT cols [0, 512)
  groupnorm_kernel<<<BB * GROUPS, 256, 0, stream>>>(query, qn, cnT, 0, norm_w, norm_b);
  // 3. key projection -> kraw
  {
    int waves = BB * (QC / 32) * (TT / 32);
    kproj_kernel<<<waves / 8, 256, 0, stream>>>(kpw_bf, kdsT, kraw);
  }
  // 4. GroupNorm(k) -> cnT cols [512, 1024)
  groupnorm_kernel<<<BB * GROUPS, 256, 0, stream>>>(kraw, nullptr, cnT, QC, norm_w, norm_b);
  // 5. qkv projection
  {
    int waves = BB * (3 * QC / 32) * (TT / 32);
    qkv_kernel<<<waves / 8, 256, 0, stream>>>(qkvw_bf, qkvb, cnT, qqT, kkT, vv);
  }
  // 6-8. attention core, per batch (reuses logits/probs scratch)
  for (int b = 0; b < BB; ++b) {
    {
      int waves = (TT / 64) * (TT / 64);   // 4096 waves
      logits_kernel<<<waves / 8, 256, 0, stream>>>(b, qqT, kkT, Wm, logits);
    }
    softmax_kernel<<<TT, 256, 0, stream>>>(logits, probs);
    {
      int waves = (QC / 64) * (TT / 64);   // 512 waves
      av_kernel<<<waves / 8, 256, 0, stream>>>(b, vv, probs, abfT);
    }
  }
  // 9. output projection + bias + residual
  {
    int waves = BB * (QC / 32) * (TT / 32);
    proj_kernel<<<waves / 8, 256, 0, stream>>>(pw_bf, pb, abfT, qn, out);
  }
}
